// Masker_69947837383272
// MI455X (gfx1250) — matrix operand order repaired and checked by reading, not compiled, not run on hardware
//
#include <hip/hip_runtime.h>

#define NNODES 10000
#define NEDGES 160000
#define NP     10016            // NNODES padded to multiple of 32
#define EPAD   (NEDGES + NNODES)
#define HS     8                // head stride (pad H to 8)

typedef __bf16 bf16_t;
typedef __attribute__((ext_vector_type(16))) __bf16 v16bf;
typedef __attribute__((ext_vector_type(8)))  float  v8f;

// ---------------------------------------------------------------- utilities
__device__ inline void atomicMaxF(float* addr, float val) {
    unsigned int* ua = (unsigned int*)addr;
    unsigned int old = *ua;
    while (__uint_as_float(old) < val) {
        unsigned int assumed = old;
        old = atomicCAS(ua, assumed, __float_as_uint(val));
        if (old == assumed) break;
    }
}

// ------------------------------------------------- pad + fp32 -> bf16 of x
__global__ void k_pad_convert(const float* __restrict__ x, bf16_t* __restrict__ hb, int cols) {
    size_t idx = (size_t)blockIdx.x * blockDim.x + threadIdx.x;
    size_t total = (size_t)NP * cols;
    if (idx >= total) return;
    int n = (int)(idx / cols);
    float v = (n < NNODES) ? x[idx] : 0.0f;
    hb[idx] = (bf16_t)v;
}

// ------------------------------------ weight [K,M] fp32 -> [M,K] bf16 (B^T)
__global__ void k_wtrans(const float* __restrict__ W, bf16_t* __restrict__ Wt, int K, int M) {
    size_t idx = (size_t)blockIdx.x * blockDim.x + threadIdx.x;
    if (idx >= (size_t)K * M) return;
    int m = (int)(idx / K);
    int k = (int)(idx % K);
    Wt[idx] = (bf16_t)W[(size_t)k * M + m];
}

// --------------------------------------------------------- WMMA bf16 GEMM
// C[NP,M] = A[NP,K](bf16,row-major) * Bt[M,K](bf16,row-major == B^T)
// block = 128 threads = 4 waves; each wave: 32x32 tile; block: 32 rows x 128 cols
__global__ void __launch_bounds__(128)
k_gemm_bf16(const bf16_t* __restrict__ A, const bf16_t* __restrict__ Bt,
            float* __restrict__ C, int K, int M) {
    const int lane = threadIdx.x & 31;
    const int wave = threadIdx.x >> 5;
    const int lrow = lane & 15;
    const int akg  = (lane >> 4) << 3;   // A: k group 0 or 8
    const int bkg  = (lane >> 4) << 3;   // B: k group 0 or 8
    const int r = blockIdx.y << 5;
    const int c = (blockIdx.x << 7) + (wave << 5);

    v8f acc00 = {}, acc01 = {}, acc10 = {}, acc11 = {};

    const bf16_t* pa = A  + (size_t)(r + lrow) * K + akg;
    const bf16_t* pb = Bt + (size_t)(c + lrow) * K + bkg;
    const size_t rowskip = (size_t)16 * K;

    for (int k0 = 0; k0 < K; k0 += 32) {
        const bf16_t* qa = pa + k0;
        const bf16_t* qb = pb + k0;
        v16bf a0, a1, b0, b1;
        // A fragment: K = {akg..akg+7} and {akg+16..akg+23}
        ((float4*)&a0)[0] = *(const float4*)(qa);
        ((float4*)&a0)[1] = *(const float4*)(qa + 16);
        ((float4*)&a1)[0] = *(const float4*)(qa + rowskip);
        ((float4*)&a1)[1] = *(const float4*)(qa + rowskip + 16);
        // B fragment: K = {bkg..bkg+7} and {bkg+16..bkg+23}
        ((float4*)&b0)[0] = *(const float4*)(qb);
        ((float4*)&b0)[1] = *(const float4*)(qb + 16);
        ((float4*)&b1)[0] = *(const float4*)(qb + rowskip);
        ((float4*)&b1)[1] = *(const float4*)(qb + rowskip + 16);

        acc00 = __builtin_amdgcn_wmma_f32_16x16x32_bf16(false, a0, false, b0, (short)0, acc00, false, false);
        acc01 = __builtin_amdgcn_wmma_f32_16x16x32_bf16(false, a0, false, b1, (short)0, acc01, false, false);
        acc10 = __builtin_amdgcn_wmma_f32_16x16x32_bf16(false, a1, false, b0, (short)0, acc10, false, false);
        acc11 = __builtin_amdgcn_wmma_f32_16x16x32_bf16(false, a1, false, b1, (short)0, acc11, false, false);
    }

    const int crow = (lane >> 4) << 3;   // +8 rows for lanes 16..31
    const int ccol = lane & 15;
    float* cp = C + (size_t)(r + crow) * M + c + ccol;
#pragma unroll
    for (int v = 0; v < 8; ++v) {
        cp[(size_t)v * M]              = acc00[v];
        cp[(size_t)v * M + 16]         = acc01[v];
        cp[((size_t)v + 16) * M]       = acc10[v];
        cp[((size_t)v + 16) * M + 16]  = acc11[v];
    }
}

// ------------------------------------------- attention scores a_s, a_d [N,H]
__global__ void k_att(const float* __restrict__ g, const float* __restrict__ asrc,
                      const float* __restrict__ adst, float* __restrict__ as_,
                      float* __restrict__ ad_, int H) {
    int idx = blockIdx.x * blockDim.x + threadIdx.x;
    if (idx >= NP * H) return;
    int n = idx / H, h = idx % H;
    const float4* gp = (const float4*)(g + (size_t)n * H * 128 + (size_t)h * 128);
    const float4* sp = (const float4*)(asrc + (size_t)h * 128);
    const float4* dp = (const float4*)(adst + (size_t)h * 128);
    float s = 0.f, d = 0.f;
#pragma unroll 4
    for (int i = 0; i < 32; ++i) {
        float4 gv = gp[i], sv = sp[i], dv = dp[i];
        s += gv.x * sv.x + gv.y * sv.y + gv.z * sv.z + gv.w * sv.w;
        d += gv.x * dv.x + gv.y * dv.y + gv.z * dv.z + gv.w * dv.w;
    }
    as_[(size_t)n * HS + h] = s;
    ad_[(size_t)n * HS + h] = d;
}

// ----------------------------------------------- init agg=0, max=-inf, den=0
__global__ void k_init(float* __restrict__ agg, float* __restrict__ mb,
                       float* __restrict__ db, size_t aggN) {
    size_t idx = (size_t)blockIdx.x * blockDim.x + threadIdx.x;
    if (idx < (size_t)NP * HS) { mb[idx] = -3.0e38f; db[idx] = 0.f; }
    if (idx < aggN) agg[idx] = 0.f;
}

// --------------------------- e = leaky_relu(a_s[src]+a_d[dst]); segment max
__global__ void k_edge_logits(const int* __restrict__ ei, const float* __restrict__ as_,
                              const float* __restrict__ ad_, float* __restrict__ ew,
                              float* __restrict__ mb, int H) {
    int e = blockIdx.x * blockDim.x + threadIdx.x;
    if (e >= EPAD) return;
    int s, d;
    if (e < NEDGES) { s = ei[e]; d = ei[NEDGES + e]; } else { s = d = e - NEDGES; }
    for (int h = 0; h < H; ++h) {
        float v = as_[(size_t)s * HS + h] + ad_[(size_t)d * HS + h];
        v = v > 0.f ? v : 0.2f * v;
        ew[(size_t)e * HS + h] = v;
        atomicMaxF(&mb[(size_t)d * HS + h], v);
    }
}

// ----------------------------------- w = exp(e - m[dst]); segment sum denom
__global__ void k_edge_exp(const int* __restrict__ ei, float* __restrict__ ew,
                           const float* __restrict__ mb, float* __restrict__ db, int H) {
    int e = blockIdx.x * blockDim.x + threadIdx.x;
    if (e >= EPAD) return;
    int d;
    if (e < NEDGES) d = ei[NEDGES + e]; else d = e - NEDGES;
    for (int h = 0; h < H; ++h) {
        float wv = __expf(ew[(size_t)e * HS + h] - mb[(size_t)d * HS + h]);
        ew[(size_t)e * HS + h] = wv;
        atomicAdd(&db[(size_t)d * HS + h], wv);
    }
}

// --------------------------- agg[dst] += h[src] * alpha   (block per edge)
__global__ void __launch_bounds__(128)
k_edge_agg(const int* __restrict__ ei, const float* __restrict__ ew,
           const float* __restrict__ db, const float* __restrict__ g,
           float* __restrict__ agg, int H, int M) {
    __shared__ float alpha[HS];
    int e = blockIdx.x;
    int t = threadIdx.x;
    int s, d;
    if (e < NEDGES) { s = ei[e]; d = ei[NEDGES + e]; } else { s = d = e - NEDGES; }
    if (t < H)
        alpha[t] = ew[(size_t)e * HS + t] / (db[(size_t)d * HS + t] + 1e-16f);
    __syncthreads();
    int c = t;  // 0..127
    for (int h = 0; h < H; ++h) {
        atomicAdd(&agg[(size_t)d * M + (size_t)h * 128 + c],
                  g[(size_t)s * M + (size_t)h * 128 + c] * alpha[h]);
    }
}

// ---------------- concat layers: elu(agg + b + lin + lin_b) -> f32 + bf16
__global__ void k_combine_cat(const float* __restrict__ agg, const float* __restrict__ bg,
                              const float* __restrict__ l, const float* __restrict__ lb,
                              float* __restrict__ hout, bf16_t* __restrict__ hbout, int M) {
    size_t idx = (size_t)blockIdx.x * blockDim.x + threadIdx.x;
    if (idx >= (size_t)NP * M) return;
    int n = (int)(idx / M), j = (int)(idx % M);
    float v = 0.f;
    if (n < NNODES) {
        v = agg[idx] + bg[j] + l[idx] + lb[j];
        v = v > 0.f ? v : (__expf(v) - 1.f);   // ELU(alpha=1)
    }
    hout[idx] = v;
    hbout[idx] = (bf16_t)v;
}

// ---------------- layer 3: mean over 6 heads + b3 + lin + lin_b (no ELU)
__global__ void k_combine_mean(const float* __restrict__ agg, const float* __restrict__ b3,
                               const float* __restrict__ l, const float* __restrict__ lb,
                               float* __restrict__ hout) {
    size_t idx = (size_t)blockIdx.x * blockDim.x + threadIdx.x;
    if (idx >= (size_t)NP * 128) return;
    int n = (int)(idx / 128), c = (int)(idx % 128);
    float v = 0.f;
    if (n < NNODES) {
        float sum = 0.f;
#pragma unroll
        for (int h = 0; h < 6; ++h) sum += agg[(size_t)n * 768 + (size_t)h * 128 + c];
        v = sum * (1.f / 6.f) + b3[c] + l[(size_t)n * 128 + c] + lb[c];
    }
    hout[idx] = v;
}

// ------------------------- final edge MLP: sigmoid([h[r];h[c]] @ w + b)
__global__ void k_mlp(const int* __restrict__ ei, const float* __restrict__ h,
                      const float* __restrict__ w, const float* __restrict__ b,
                      float* __restrict__ out) {
    int e = blockIdx.x * blockDim.x + threadIdx.x;
    if (e >= NEDGES) return;
    int r = ei[e], c = ei[NEDGES + e];
    const float4* hr = (const float4*)(h + (size_t)r * 128);
    const float4* hc = (const float4*)(h + (size_t)c * 128);
    const float4* w0 = (const float4*)(w);
    const float4* w1 = (const float4*)(w + 128);
    float acc = b[0];
#pragma unroll 4
    for (int i = 0; i < 32; ++i) {
        float4 a = hr[i], bb = hc[i], x0 = w0[i], x1 = w1[i];
        acc += a.x * x0.x + a.y * x0.y + a.z * x0.z + a.w * x0.w;
        acc += bb.x * x1.x + bb.y * x1.y + bb.z * x1.z + bb.w * x1.w;
    }
    out[e] = 1.f / (1.f + __expf(-acc));
}

// =========================================================================
extern "C" void kernel_launch(void* const* d_in, const int* in_sizes, int n_in,
                              void* d_out, int out_size, void* d_ws, size_t ws_size,
                              hipStream_t stream) {
    (void)in_sizes; (void)n_in; (void)out_size; (void)ws_size;
    const float* x     = (const float*)d_in[0];
    const int*   ei    = (const int*)  d_in[1];
    const float* W1    = (const float*)d_in[2];
    const float* asrc1 = (const float*)d_in[3];
    const float* adst1 = (const float*)d_in[4];
    const float* b1    = (const float*)d_in[5];
    const float* l1w   = (const float*)d_in[6];
    const float* l1b   = (const float*)d_in[7];
    const float* W2    = (const float*)d_in[8];
    const float* asrc2 = (const float*)d_in[9];
    const float* adst2 = (const float*)d_in[10];
    const float* b2    = (const float*)d_in[11];
    const float* l2w   = (const float*)d_in[12];
    const float* l2b   = (const float*)d_in[13];
    const float* W3    = (const float*)d_in[14];
    const float* asrc3 = (const float*)d_in[15];
    const float* adst3 = (const float*)d_in[16];
    const float* b3    = (const float*)d_in[17];
    const float* l3w   = (const float*)d_in[18];
    const float* l3b   = (const float*)d_in[19];
    const float* mlpw  = (const float*)d_in[20];
    const float* mlpb  = (const float*)d_in[21];
    float* out = (float*)d_out;

    char* ws = (char*)d_ws;
    size_t off = 0;
    auto alloc = [&](size_t bytes) -> char* {
        char* p = ws + off;
        off = (off + bytes + 255) & ~(size_t)255;
        return p;
    };
    bf16_t* hb    = (bf16_t*)alloc((size_t)NP * 512 * 2); // bf16 activations
    float*  g     = (float*) alloc((size_t)NP * 768 * 4); // GAT gemm out
    float*  lbuf  = (float*) alloc((size_t)NP * 512 * 4); // lin gemm out
    float*  agg   = (float*) alloc((size_t)NP * 768 * 4); // segment-sum out
    float*  hf    = (float*) alloc((size_t)NP * 512 * 4); // f32 activations
    float*  as_   = (float*) alloc((size_t)NP * HS * 4);
    float*  ad_   = (float*) alloc((size_t)NP * HS * 4);
    float*  mb    = (float*) alloc((size_t)NP * HS * 4);
    float*  db    = (float*) alloc((size_t)EPAD ? (size_t)NP * HS * 4 : 0);
    float*  ew    = (float*) alloc((size_t)EPAD * HS * 4);
    bf16_t* w1t   = (bf16_t*)alloc((size_t)512 * 128 * 2);
    bf16_t* l1t   = (bf16_t*)alloc((size_t)512 * 128 * 2);
    bf16_t* w2t   = (bf16_t*)alloc((size_t)512 * 512 * 2);
    bf16_t* l2t   = (bf16_t*)alloc((size_t)512 * 512 * 2);
    bf16_t* w3t   = (bf16_t*)alloc((size_t)768 * 512 * 2);
    bf16_t* l3t   = (bf16_t*)alloc((size_t)128 * 512 * 2);

    const int TB = 256;
    auto cdiv = [](size_t a, size_t b) { return (unsigned)((a + b - 1) / b); };

    // input conversion + weight transpose/convert (small, once per call)
    k_pad_convert<<<cdiv((size_t)NP * 128, TB), TB, 0, stream>>>(x, hb, 128);
    k_wtrans<<<cdiv((size_t)512 * 128, TB), TB, 0, stream>>>(W1,  w1t, 128, 512);
    k_wtrans<<<cdiv((size_t)512 * 128, TB), TB, 0, stream>>>(l1w, l1t, 128, 512);
    k_wtrans<<<cdiv((size_t)512 * 512, TB), TB, 0, stream>>>(W2,  w2t, 512, 512);
    k_wtrans<<<cdiv((size_t)512 * 512, TB), TB, 0, stream>>>(l2w, l2t, 512, 512);
    k_wtrans<<<cdiv((size_t)768 * 512, TB), TB, 0, stream>>>(W3,  w3t, 512, 768);
    k_wtrans<<<cdiv((size_t)128 * 512, TB), TB, 0, stream>>>(l3w, l3t, 512, 128);

    auto run_gat = [&](const bf16_t* act, int K, const bf16_t* wt, int Mg,
                       const bf16_t* lt, int Ml,
                       const float* asrc, const float* adst, int H) {
        dim3 gg(Mg / 128, NP / 32);
        k_gemm_bf16<<<gg, 128, 0, stream>>>(act, wt, g, K, Mg);
        dim3 gl(Ml / 128, NP / 32);
        k_gemm_bf16<<<gl, 128, 0, stream>>>(act, lt, lbuf, K, Ml);
        k_att<<<cdiv((size_t)NP * H, TB), TB, 0, stream>>>(g, asrc, adst, as_, ad_, H);
        size_t aggN = (size_t)NP * Mg;
        k_init<<<cdiv(aggN, TB), TB, 0, stream>>>(agg, mb, db, aggN);
        k_edge_logits<<<cdiv((size_t)EPAD, TB), TB, 0, stream>>>(ei, as_, ad_, ew, mb, H);
        k_edge_exp<<<cdiv((size_t)EPAD, TB), TB, 0, stream>>>(ei, ew, mb, db, H);
        k_edge_agg<<<EPAD, 128, 0, stream>>>(ei, ew, db, g, agg, H, Mg);
    };

    // layer 1: in 128 -> out 512 (4 heads, concat)
    run_gat(hb, 128, w1t, 512, l1t, 512, asrc1, adst1, 4);
    k_combine_cat<<<cdiv((size_t)NP * 512, TB), TB, 0, stream>>>(agg, b1, lbuf, l1b, hf, hb, 512);

    // layer 2: 512 -> 512 (4 heads, concat)
    run_gat(hb, 512, w2t, 512, l2t, 512, asrc2, adst2, 4);
    k_combine_cat<<<cdiv((size_t)NP * 512, TB), TB, 0, stream>>>(agg, b2, lbuf, l2b, hf, hb, 512);

    // layer 3: 512 -> 128 (6 heads, mean)
    run_gat(hb, 512, w3t, 768, l3t, 128, asrc3, adst3, 6);
    k_combine_mean<<<cdiv((size_t)NP * 128, TB), TB, 0, stream>>>(agg, b3, lbuf, l3b, hf);

    // edge scorer
    k_mlp<<<cdiv((size_t)NEDGES, TB), TB, 0, stream>>>(ei, hf, mlpw, mlpb, out);
}
